// EncodeLayer_3289944949177
// MI455X (gfx1250) — compile-verified, matrix operand order checked by reading, not run on hardware
//
#include <hip/hip_runtime.h>
#include <hip/hip_bf16.h>
#include <stdint.h>

// ---------------------------------------------------------------------------
// CDNA5 / gfx1250 bf16 WMMA transformer encoder layer.
// Block tile 64x64 (8 waves, each wave = 16x32 -> 2x v_wmma_f32_16x16x32_bf16
// per K-step sharing one A fragment), ping-pong LDS double buffering (one
// barrier per K-step), global_prefetch_b8 lookahead, NT stores for the 537MB
// attention tensor.
// ---------------------------------------------------------------------------

typedef __attribute__((ext_vector_type(16))) __bf16       v16bf;
typedef __attribute__((ext_vector_type(8)))  float        v8f;
typedef __attribute__((ext_vector_type(8)))  unsigned int v8u;

#define LDS_STR 40   // bf16 elems per LDS row (32 data + 8 pad; keeps b128 merges)

__device__ __forceinline__ unsigned short f2bf(float f) {
  unsigned int u = __float_as_uint(f);
  u += 0x7FFFu + ((u >> 16) & 1u);            // round-to-nearest-even
  return (unsigned short)(u >> 16);
}
__device__ __forceinline__ unsigned int pk2(float lo, float hi) {
  return (unsigned int)f2bf(lo) | ((unsigned int)f2bf(hi) << 16);
}
// order-preserving uint encoding of float (for atomic global min)
__device__ __forceinline__ unsigned int ford(float f) {
  unsigned int u = __float_as_uint(f);
  return (u & 0x80000000u) ? ~u : (u | 0x80000000u);
}

// 16x32 bf16 A/B fragment from an LDS tile row. ISA 16-bit layout:
// VGPR p<4 -> K = half*8 + 2p ; p>=4 -> K = 16 + half*8 + 2(p-4).
// Contiguous per quad -> compiler merges into 2x ds_load_b128.
__device__ __forceinline__ v8u load_frag(const unsigned short* lds, int row, int half) {
  v8u u;
  const unsigned short* base = lds + row * LDS_STR;
#pragma unroll
  for (int p = 0; p < 8; ++p) {
    int kk = (p < 4) ? (half * 8 + 2 * p) : (16 + half * 8 + 2 * (p - 4));
    u[p] = *(const unsigned int*)(base + kk);
  }
  return u;
}

__device__ __forceinline__ v8f wmma_bf16(v8u a, v8u b, v8f c) {
  return __builtin_amdgcn_wmma_f32_16x16x32_bf16(
      false, __builtin_bit_cast(v16bf, a),
      false, __builtin_bit_cast(v16bf, b),
      (short)0, c, false, false);
}

// store 8 f32 (2x float4) as 4 packed-bf16 dwords into an LDS tile row
__device__ __forceinline__ void st_tile_f32(unsigned short* lds, int r, int c0,
                                            float4 x, float4 y) {
  unsigned int* d = (unsigned int*)(lds + r * LDS_STR + c0);
  d[0] = pk2(x.x, x.y); d[1] = pk2(x.z, x.w);
  d[2] = pk2(y.x, y.y); d[3] = pk2(y.z, y.w);
}
__device__ __forceinline__ void st_tile_bf16(unsigned short* lds, int r, int c0, uint4 v) {
  unsigned int* d = (unsigned int*)(lds + r * LDS_STR + c0);
  d[0] = v.x; d[1] = v.y; d[2] = v.z; d[3] = v.w;
}

// ---------------------------------------------------------------------------
// Projection: C = X @ W^T, X[8192,1024] f32, W[1024,1024] f32 -> bf16 out,
// head-split [b,h,t,d] (TRANS=0) or [b,h,d,t] (TRANS=1), scaled.
// ---------------------------------------------------------------------------
template <int TRANS>
__global__ void k_proj(const float* __restrict__ X, const float* __restrict__ W,
                       unsigned short* __restrict__ dst, float scale) {
  __shared__ unsigned short As[2][64 * LDS_STR];
  __shared__ unsigned short Bs[2][64 * LDS_STR];
  const int tid = threadIdx.x;
  const int lane = tid & 31, wid = tid >> 5;
  const int wm = wid >> 1, wn = wid & 1;
  const int half = lane >> 4, mrow = lane & 15;
  const int m0 = blockIdx.x * 64;
  const int n0 = blockIdx.y * 64;
  const int lr = tid >> 2, lc = (tid & 3) * 8;
  const float* ap = X + (size_t)(m0 + lr) * 1024 + lc;
  const float* bp = W + (size_t)(n0 + lr) * 1024 + lc;

  st_tile_f32(As[0], lr, lc, *(const float4*)ap, *(const float4*)(ap + 4));
  st_tile_f32(Bs[0], lr, lc, *(const float4*)bp, *(const float4*)(bp + 4));
  __syncthreads();

  v8f acc0 = {}, acc1 = {};
  int buf = 0;
  for (int k0 = 0; k0 < 1024; k0 += 32) {
    const bool more = (k0 + 32) < 1024;
    float4 a0, a1, b0, b1;
    if (more) {
      a0 = *(const float4*)(ap + k0 + 32); a1 = *(const float4*)(ap + k0 + 36);
      b0 = *(const float4*)(bp + k0 + 32); b1 = *(const float4*)(bp + k0 + 36);
      if (k0 + 64 < 1024) {
        __builtin_prefetch(ap + k0 + 64, 0, 3);
        __builtin_prefetch(bp + k0 + 64, 0, 3);
      }
    }
    v8u au  = load_frag(As[buf], wm * 16 + mrow, half);
    v8u bu0 = load_frag(Bs[buf], wn * 32 + mrow, half);
    v8u bu1 = load_frag(Bs[buf], wn * 32 + 16 + mrow, half);
    acc0 = wmma_bf16(au, bu0, acc0);
    acc1 = wmma_bf16(au, bu1, acc1);
    if (more) {
      st_tile_f32(As[buf ^ 1], lr, lc, a0, a1);
      st_tile_f32(Bs[buf ^ 1], lr, lc, b0, b1);
    }
    __syncthreads();
    buf ^= 1;
  }
#pragma unroll
  for (int r = 0; r < 8; ++r) {
    int gM = m0 + wm * 16 + half * 8 + r;
    int b = gM >> 10, t = gM & 1023;
#pragma unroll
    for (int s = 0; s < 2; ++s) {
      int gN = n0 + wn * 32 + s * 16 + mrow;
      int h = gN >> 6, d = gN & 63;
      size_t idx = TRANS ? ((size_t)(b * 16 + h) * 64 + d) * 1024 + t
                         : ((size_t)(b * 16 + h) * 1024 + t) * 64 + d;
      dst[idx] = f2bf((s ? acc1[r] : acc0[r]) * scale);
    }
  }
}

// ---------------------------------------------------------------------------
// Scores: S[bh] = qh[bh] @ kh[bh]^T (both [1024,64] bf16) -> f32 to d_out attn
// region (NT stores) with fused order-preserving atomic global min.
// ---------------------------------------------------------------------------
__global__ void k_scores(const unsigned short* __restrict__ qh,
                         const unsigned short* __restrict__ kh,
                         float* __restrict__ S, unsigned int* __restrict__ gmin) {
  __shared__ unsigned short As[2][64 * LDS_STR];
  __shared__ unsigned short Bs[2][64 * LDS_STR];
  __shared__ float red[256];
  const int tid = threadIdx.x;
  const int lane = tid & 31, wid = tid >> 5;
  const int wm = wid >> 1, wn = wid & 1;
  const int half = lane >> 4, mrow = lane & 15;
  const int bh = blockIdx.z;
  const int q0 = blockIdx.x * 64;
  const int n0 = blockIdx.y * 64;
  const int lr = tid >> 2, lc = (tid & 3) * 8;
  const unsigned int* qp = (const unsigned int*)(qh + (size_t)bh * 1024 * 64)
                           + ((size_t)(q0 + lr) * 64 + lc) / 2;
  const unsigned int* kp = (const unsigned int*)(kh + (size_t)bh * 1024 * 64)
                           + ((size_t)(n0 + lr) * 64 + lc) / 2;

  st_tile_bf16(As[0], lr, lc, *(const uint4*)qp);
  st_tile_bf16(Bs[0], lr, lc, *(const uint4*)kp);
  __syncthreads();

  v8f acc0 = {}, acc1 = {};
  int buf = 0;
#pragma unroll
  for (int k0 = 0; k0 < 64; k0 += 32) {
    const bool more = (k0 + 32) < 64;
    uint4 a4, b4;
    if (more) { a4 = *(const uint4*)(qp + 16); b4 = *(const uint4*)(kp + 16); }
    v8u au  = load_frag(As[buf], wm * 16 + mrow, half);
    v8u bu0 = load_frag(Bs[buf], wn * 32 + mrow, half);
    v8u bu1 = load_frag(Bs[buf], wn * 32 + 16 + mrow, half);
    acc0 = wmma_bf16(au, bu0, acc0);
    acc1 = wmma_bf16(au, bu1, acc1);
    if (more) {
      st_tile_bf16(As[buf ^ 1], lr, lc, a4);
      st_tile_bf16(Bs[buf ^ 1], lr, lc, b4);
    }
    __syncthreads();
    buf ^= 1;
  }
  float lmin = acc0[0];
#pragma unroll
  for (int r = 0; r < 8; ++r) {
    int gq = q0 + wm * 16 + half * 8 + r;
    size_t rowb = ((size_t)bh * 1024 + gq) * 1024;
    int gk0 = n0 + wn * 32 + mrow;
    __builtin_nontemporal_store(acc0[r], &S[rowb + gk0]);
    __builtin_nontemporal_store(acc1[r], &S[rowb + gk0 + 16]);
    lmin = fminf(lmin, fminf(acc0[r], acc1[r]));
  }
  red[tid] = lmin;
  __syncthreads();
  for (int s = 128; s > 0; s >>= 1) {
    if (tid < s) red[tid] = fminf(red[tid], red[tid + s]);
    __syncthreads();
  }
  if (tid == 0) atomicMin(gmin, ford(red[0]));
}

__global__ void k_init(unsigned int* gmin) { *gmin = 0xFFFFFFFFu; }

__global__ void k_minfin(const unsigned int* __restrict__ gmin, float* __restrict__ inv) {
  unsigned int ou = *gmin;
  unsigned int u = (ou & 0x80000000u) ? (ou & 0x7FFFFFFFu) : ~ou;
  *inv = 1.0f / fabsf(__uint_as_float(u));
}

// ---------------------------------------------------------------------------
// In-place per-row: scale by 1/|min|, L2-normalize (eps=1e-12), softmax.
// ---------------------------------------------------------------------------
__global__ void k_rowsm(float* __restrict__ S, const float* __restrict__ inv) {
  __shared__ float red[256];
  const int tid = threadIdx.x;
  const size_t row = (size_t)blockIdx.y * 1024 + blockIdx.x;
  float* p = S + row * 1024;
  const float sc = *inv;
  float x[4];
  float ss = 0.f;
#pragma unroll
  for (int j = 0; j < 4; ++j) { x[j] = p[tid + 256 * j] * sc; ss += x[j] * x[j]; }
  red[tid] = ss; __syncthreads();
  for (int s = 128; s > 0; s >>= 1) { if (tid < s) red[tid] += red[tid + s]; __syncthreads(); }
  float rden = 1.0f / fmaxf(sqrtf(red[0]), 1e-12f);
  __syncthreads();
  float mx = -3.402823466e38f;
#pragma unroll
  for (int j = 0; j < 4; ++j) { x[j] *= rden; mx = fmaxf(mx, x[j]); }
  red[tid] = mx; __syncthreads();
  for (int s = 128; s > 0; s >>= 1) { if (tid < s) red[tid] = fmaxf(red[tid], red[tid + s]); __syncthreads(); }
  mx = red[0];
  __syncthreads();
  float es = 0.f;
#pragma unroll
  for (int j = 0; j < 4; ++j) { x[j] = __expf(x[j] - mx); es += x[j]; }
  red[tid] = es; __syncthreads();
  for (int s = 128; s > 0; s >>= 1) { if (tid < s) red[tid] += red[tid + s]; __syncthreads(); }
  float rs = 1.0f / red[0];
#pragma unroll
  for (int j = 0; j < 4; ++j) __builtin_nontemporal_store(x[j] * rs, &p[tid + 256 * j]);
}

// ---------------------------------------------------------------------------
// O[bh] = softmax(S)[1024,1024] @ V[bh]; V in [b,h,d,t] so the B-tile reduction
// dim is contiguous. Output bf16 merged [b, t, h*64 + d].
// ---------------------------------------------------------------------------
__global__ void k_attnv(const float* __restrict__ S, const unsigned short* __restrict__ vhT,
                        unsigned short* __restrict__ outh) {
  __shared__ unsigned short As[2][64 * LDS_STR];
  __shared__ unsigned short Bs[2][64 * LDS_STR];
  const int tid = threadIdx.x;
  const int lane = tid & 31, wid = tid >> 5;
  const int wm = wid >> 1, wn = wid & 1;
  const int half = lane >> 4, mrow = lane & 15;
  const int bh = blockIdx.z;
  const int b = bh >> 4, h = bh & 15;
  const int t0 = blockIdx.x * 64;
  const int lr = tid >> 2, lc = (tid & 3) * 8;
  const float* ap = S + (size_t)bh * 1024 * 1024 + (size_t)(t0 + lr) * 1024 + lc;
  const unsigned int* vp = (const unsigned int*)(vhT + (size_t)bh * 64 * 1024)
                           + ((size_t)lr * 1024 + lc) / 2;

  st_tile_f32(As[0], lr, lc, *(const float4*)ap, *(const float4*)(ap + 4));
  st_tile_bf16(Bs[0], lr, lc, *(const uint4*)vp);
  __syncthreads();

  v8f acc0 = {}, acc1 = {};
  int buf = 0;
  for (int k0 = 0; k0 < 1024; k0 += 32) {
    const bool more = (k0 + 32) < 1024;
    float4 a0, a1; uint4 b4;
    if (more) {
      a0 = *(const float4*)(ap + k0 + 32); a1 = *(const float4*)(ap + k0 + 36);
      b4 = *(const uint4*)(vp + (k0 + 32) / 2);
      if (k0 + 64 < 1024) {
        __builtin_prefetch(ap + k0 + 64, 0, 3);
        __builtin_prefetch(vp + (k0 + 64) / 2, 0, 3);
      }
    }
    v8u au  = load_frag(As[buf], wm * 16 + mrow, half);
    v8u bu0 = load_frag(Bs[buf], wn * 32 + mrow, half);
    v8u bu1 = load_frag(Bs[buf], wn * 32 + 16 + mrow, half);
    acc0 = wmma_bf16(au, bu0, acc0);
    acc1 = wmma_bf16(au, bu1, acc1);
    if (more) {
      st_tile_f32(As[buf ^ 1], lr, lc, a0, a1);
      st_tile_bf16(Bs[buf ^ 1], lr, lc, b4);
    }
    __syncthreads();
    buf ^= 1;
  }
#pragma unroll
  for (int r = 0; r < 8; ++r) {
    int gt = t0 + wm * 16 + half * 8 + r;
    size_t rowb = (size_t)(b * 1024 + gt) * 1024 + h * 64;
    int gd0 = wn * 32 + mrow;
    outh[rowb + gd0]      = f2bf(acc0[r]);
    outh[rowb + gd0 + 16] = f2bf(acc1[r]);
  }
}

// ---------------------------------------------------------------------------
// FC: out = H @ Wfc^T + residual ; H[8192,1024] bf16, Wfc[1024,1024] f32.
// ---------------------------------------------------------------------------
__global__ void k_fc(const unsigned short* __restrict__ H, const float* __restrict__ Wfc,
                     const float* __restrict__ resid, float* __restrict__ out) {
  __shared__ unsigned short As[2][64 * LDS_STR];
  __shared__ unsigned short Bs[2][64 * LDS_STR];
  const int tid = threadIdx.x;
  const int lane = tid & 31, wid = tid >> 5;
  const int wm = wid >> 1, wn = wid & 1;
  const int half = lane >> 4, mrow = lane & 15;
  const int m0 = blockIdx.x * 64;
  const int n0 = blockIdx.y * 64;
  const int lr = tid >> 2, lc = (tid & 3) * 8;
  const unsigned int* hp = (const unsigned int*)H + ((size_t)(m0 + lr) * 1024 + lc) / 2;
  const float* bp = Wfc + (size_t)(n0 + lr) * 1024 + lc;

  st_tile_bf16(As[0], lr, lc, *(const uint4*)hp);
  st_tile_f32(Bs[0], lr, lc, *(const float4*)bp, *(const float4*)(bp + 4));
  __syncthreads();

  v8f acc0 = {}, acc1 = {};
  int buf = 0;
  for (int k0 = 0; k0 < 1024; k0 += 32) {
    const bool more = (k0 + 32) < 1024;
    uint4 a4; float4 b0, b1;
    if (more) {
      a4 = *(const uint4*)(hp + (k0 + 32) / 2);
      b0 = *(const float4*)(bp + k0 + 32); b1 = *(const float4*)(bp + k0 + 36);
      if (k0 + 64 < 1024) {
        __builtin_prefetch(hp + (k0 + 64) / 2, 0, 3);
        __builtin_prefetch(bp + k0 + 64, 0, 3);
      }
    }
    v8u au  = load_frag(As[buf], wm * 16 + mrow, half);
    v8u bu0 = load_frag(Bs[buf], wn * 32 + mrow, half);
    v8u bu1 = load_frag(Bs[buf], wn * 32 + 16 + mrow, half);
    acc0 = wmma_bf16(au, bu0, acc0);
    acc1 = wmma_bf16(au, bu1, acc1);
    if (more) {
      st_tile_bf16(As[buf ^ 1], lr, lc, a4);
      st_tile_f32(Bs[buf ^ 1], lr, lc, b0, b1);
    }
    __syncthreads();
    buf ^= 1;
  }
#pragma unroll
  for (int r = 0; r < 8; ++r) {
    int gM = m0 + wm * 16 + half * 8 + r;
    int gN = n0 + wn * 32 + mrow;
    size_t i0 = (size_t)gM * 1024 + gN;
    out[i0]      = acc0[r] + resid[i0];
    out[i0 + 16] = acc1[r] + resid[i0 + 16];
  }
}

// ---------------------------------------------------------------------------
// LayerNorm over last dim (1024), eps=1e-6, biased variance.
// ---------------------------------------------------------------------------
__global__ void k_ln(const float* __restrict__ X, const float* __restrict__ gamma,
                     const float* __restrict__ beta, float* __restrict__ out) {
  __shared__ float red[256];
  const int tid = threadIdx.x;
  const size_t row = blockIdx.x;
  const float* p = X + row * 1024;
  float x[4]; float s = 0.f, sq = 0.f;
#pragma unroll
  for (int j = 0; j < 4; ++j) { x[j] = p[tid + 256 * j]; s += x[j]; sq += x[j] * x[j]; }
  red[tid] = s; __syncthreads();
  for (int k = 128; k > 0; k >>= 1) { if (tid < k) red[tid] += red[tid + k]; __syncthreads(); }
  float mean = red[0] * (1.0f / 1024.0f);
  __syncthreads();
  red[tid] = sq; __syncthreads();
  for (int k = 128; k > 0; k >>= 1) { if (tid < k) red[tid] += red[tid + k]; __syncthreads(); }
  float var = red[0] * (1.0f / 1024.0f) - mean * mean;
  float rstd = rsqrtf(var + 1e-6f);
#pragma unroll
  for (int j = 0; j < 4; ++j) {
    int c = tid + 256 * j;
    out[row * 1024 + c] = (x[j] - mean) * rstd * gamma[c] + beta[c];
  }
}

// ---------------------------------------------------------------------------
extern "C" void kernel_launch(void* const* d_in, const int* in_sizes, int n_in,
                              void* d_out, int out_size, void* d_ws, size_t ws_size,
                              hipStream_t stream) {
  (void)in_sizes; (void)n_in; (void)out_size; (void)ws_size;
  const float* q    = (const float*)d_in[0];
  const float* k    = (const float*)d_in[1];
  const float* v    = (const float*)d_in[2];
  const float* Wq   = (const float*)d_in[3];
  const float* Wk   = (const float*)d_in[4];
  const float* Wv   = (const float*)d_in[5];
  const float* Wfc  = (const float*)d_in[6];
  const float* gam  = (const float*)d_in[7];
  const float* bet  = (const float*)d_in[8];

  float* out  = (float*)d_out;                 // [8192, 1024]
  float* attn = out + (size_t)8192 * 1024;     // [128, 1024, 1024] (raw scores -> softmaxed in place)

  // Workspace layout (~100.7 MB):
  const size_t H = (size_t)8192 * 1024;
  unsigned short* qh   = (unsigned short*)d_ws;      // bf16 [b,h,t,d], pre-scaled by 1/8
  unsigned short* kh   = qh + H;                     // bf16 [b,h,t,d]
  unsigned short* vhT  = kh + H;                     // bf16 [b,h,d,t]
  unsigned short* outh = vhT + H;                    // bf16 [b,t,h*d]
  float*          fcout = (float*)(outh + H);        // f32  [8192,1024] pre-LN
  unsigned int*   gmin  = (unsigned int*)(fcout + H);
  float*          invmin = (float*)(gmin + 1);

  dim3 blk(256);
  k_init     <<<1, 1, 0, stream>>>(gmin);
  k_proj<0>  <<<dim3(128, 16),     blk, 0, stream>>>(q, Wq, qh, 0.125f); // 1/sqrt(64)
  k_proj<0>  <<<dim3(128, 16),     blk, 0, stream>>>(k, Wk, kh, 1.0f);
  k_proj<1>  <<<dim3(128, 16),     blk, 0, stream>>>(v, Wv, vhT, 1.0f);
  k_scores   <<<dim3(16, 16, 128), blk, 0, stream>>>(qh, kh, attn, gmin);
  k_minfin   <<<1, 1, 0, stream>>>(gmin, invmin);
  k_rowsm    <<<dim3(1024, 128),   blk, 0, stream>>>(attn, invmin);
  k_attnv    <<<dim3(16, 1, 128),  blk, 0, stream>>>(attn, vhT, outh);
  k_fc       <<<dim3(128, 16),     blk, 0, stream>>>(outh, Wfc, q, fcout);
  k_ln       <<<8192,              blk, 0, stream>>>(fcout, gam, bet, out);
}
